// CausalLM_36910948942461
// MI455X (gfx1250) — matrix operand order repaired and checked by reading, not compiled, not run on hardware
//
#include <hip/hip_runtime.h>

// ---------------------------------------------------------------------------
// Types
// ---------------------------------------------------------------------------
typedef __attribute__((ext_vector_type(16))) __bf16 bf16x16;
typedef __attribute__((ext_vector_type(8)))  __bf16 bf16x8;
typedef __attribute__((ext_vector_type(8)))  float  floatx8;
typedef __attribute__((ext_vector_type(4)))  float  floatx4;
typedef __attribute__((ext_vector_type(4)))  unsigned int uint32x4;
typedef __attribute__((ext_vector_type(4)))  int    int32x4;
typedef __attribute__((ext_vector_type(8)))  int    int32x8;

#if __has_builtin(__builtin_amdgcn_tensor_load_to_lds)
#define ATH_HAVE_TDM 1
#else
#define ATH_HAVE_TDM 0
#endif

__device__ __forceinline__ __bf16 to_bf16(float f) {
  return (__bf16)f;  // native v_cvt_pk_bf16_f32 on gfx1250
}

__device__ __forceinline__ floatx8 wmma_bf16(bf16x16 a, bf16x16 b, floatx8 c) {
  // D = A(16x32) * B(32x16) + C, fp32 accumulate
  return __builtin_amdgcn_wmma_f32_16x16x32_bf16(false, a, false, b, (short)0, c,
                                                 false, false);
}

#if ATH_HAVE_TDM
// Issue a TDM load of a 2-D fp32 tile (tile_d1 rows x tile_d0 contiguous
// elements) from global into LDS.  D# layout per CDNA5 ISA ch. 8.
__device__ __forceinline__ void tdm_load_tile_2d(unsigned lds_addr,
                                                 const void* gaddr,
                                                 int tensor_d0, int tensor_d1,
                                                 int tile_d0, int tile_d1,
                                                 int stride0) {
  unsigned long long ga = (unsigned long long)gaddr;
  uint32x4 g0;
  g0[0] = 1u;                                      // count=1, user descriptor
  g0[1] = lds_addr;                                // lds_addr (bytes)
  g0[2] = (unsigned)ga;                            // global_addr[31:0]
  g0[3] = (unsigned)((ga >> 32) & 0x1FFFFFFull) |  // global_addr[56:32]
          (2u << 30);                              // type = 2 ("image")
  int32x8 g1;
  g1[0] = (int)(2u << 16);                         // data_size = 4 bytes
  g1[1] = (int)(((unsigned)tensor_d0 & 0xFFFFu) << 16);          // dim0[15:0]
  g1[2] = (int)(((unsigned)tensor_d0 >> 16) |
                (((unsigned)tensor_d1 & 0xFFFFu) << 16));        // dim0hi,dim1lo
  g1[3] = (int)((((unsigned)tensor_d1 >> 16) & 0xFFFFu) |
                ((unsigned)tile_d0 << 16));                      // dim1hi,tile0
  g1[4] = (int)((unsigned)tile_d1 & 0xFFFFu);      // tile_dim1 (tile_dim2 = 0)
  g1[5] = stride0;                                 // tensor_dim0_stride[31:0]
  g1[6] = 0;                                       // stride0 hi, stride1 lo
  g1[7] = 0;
  int32x4 z4 = {0, 0, 0, 0};
#if defined(__clang_major__) && __clang_major__ >= 23
  int32x8 z8 = {0, 0, 0, 0, 0, 0, 0, 0};
  __builtin_amdgcn_tensor_load_to_lds(g0, g1, z4, z4, z8, 0);
#else
  __builtin_amdgcn_tensor_load_to_lds(g0, g1, z4, z4, 0);
#endif
}
#endif  // ATH_HAVE_TDM

// ---------------------------------------------------------------------------
// Embedding gather: x[m][d] = embed[tokens[m]][d]
// ---------------------------------------------------------------------------
__global__ __launch_bounds__(256) void embed_gather_kernel(
    const int* __restrict__ tokens, const float* __restrict__ embed,
    float* __restrict__ x, int M, int D) {
  long i = (long)blockIdx.x * 256 + threadIdx.x;
  long total = (long)M * D;
  if (i >= total) return;
  int d = (int)(i % D);
  long m = i / D;
  int tok = tokens[m];
  x[i] = embed[(long)tok * D + d];
}

// ---------------------------------------------------------------------------
// RMSNorm: out[r][:] = x[r][:] * rsqrt(mean(x^2)+eps) * g[:]
// ---------------------------------------------------------------------------
__global__ __launch_bounds__(256) void rmsnorm_kernel(
    const float* __restrict__ x, const float* __restrict__ g,
    float* __restrict__ out, int D) {
  int row = blockIdx.x;
  int tid = threadIdx.x;
  int wave = tid >> 5, lane = tid & 31;
  const float* xr = x + (long)row * D;
  float ss = 0.f;
  for (int i = tid; i < D; i += 256) {
    float v = xr[i];
    ss += v * v;
  }
  for (int o = 16; o > 0; o >>= 1) ss += __shfl_xor(ss, o);
  __shared__ float red[9];
  if (lane == 0) red[wave] = ss;
  __syncthreads();
  if (tid == 0) {
    float t = 0.f;
    for (int i = 0; i < 8; ++i) t += red[i];
    red[8] = rsqrtf(t / (float)D + 1e-6f);
  }
  __syncthreads();
  float inv = red[8];
  for (int i = tid; i < D; i += 256) out[(long)row * D + i] = xr[i] * inv * g[i];
}

// ---------------------------------------------------------------------------
// RoPE (interleaved pairs), x layout [rows][Hn*64], dk = 64, dk/2 = 32
// ---------------------------------------------------------------------------
__global__ __launch_bounds__(256) void rope_kernel(
    float* __restrict__ x, const float* __restrict__ cs,
    const float* __restrict__ sn, int S, int Hn, int total) {
  int i = blockIdx.x * 256 + threadIdx.x;
  if (i >= total) return;
  int j = i & 31;
  int t = i >> 5;
  int hh = t % Hn;
  int bs = t / Hn;
  int s = bs % S;
  float c = cs[s * 32 + j];
  float si = sn[s * 32 + j];
  float* p = x + (long)bs * (Hn * 64) + hh * 64 + 2 * j;
  float xr = p[0], xi = p[1];
  p[0] = xr * c - xi * si;
  p[1] = xr * si + xi * c;
}

// ---------------------------------------------------------------------------
// SiLU gate: u1 = silu(u1) * u2
// ---------------------------------------------------------------------------
__global__ __launch_bounds__(256) void silu_mul_kernel(
    float* __restrict__ u1, const float* __restrict__ u2, int total) {
  int i = blockIdx.x * 256 + threadIdx.x;
  if (i >= total) return;
  float a = u1[i];
  float s = a / (1.f + __expf(-a));
  u1[i] = s * u2[i];
}

// ---------------------------------------------------------------------------
// Tiled WMMA GEMM:  C[M,N] = A[M,K] @ W + bias (+ residual)
//   BT=false: W is [K,N] row-major.   BT=true: W is [N,K] row-major (B^T).
// Block tile 128x128, K-step 32, 256 threads = 8 waves (wave32),
// wave grid 2x4, 64x32 per wave = 8 v_wmma per K-step.
// Software-pipelined: double-buffered LDS; the Tensor Data Mover DMAs
// tile i+1 while the waves compute on tile i; s_wait_tensorcnt sits at the
// end of the iteration so the DMA gets the whole compute window.
// ---------------------------------------------------------------------------
template <bool BT, bool HAS_BIAS, bool HAS_RES>
__global__ __launch_bounds__(256) void gemm_wmma_kernel(
    const float* __restrict__ A, const float* __restrict__ W,
    const float* __restrict__ bias, const float* __restrict__ Res,
    float* __restrict__ C, int M, int N, int K) {
  __shared__ float sAf[2][128][32];  // [buf][m][k] fp32
  __shared__ float sBf[2][128][32];  // [buf][n][k] fp32 (B stored transposed)

  const int tid = threadIdx.x;
  const int bm = blockIdx.y * 128;
  const int bn = blockIdx.x * 128;
  const int wave = tid >> 5, lane = tid & 31;
  const int wm = (wave >> 2) * 64;
  const int wn = (wave & 3) * 32;
  const int hf = lane >> 4;
  const int lrow = lane & 15;

  // manual staging coordinates
  const int arow = tid >> 1;        // 0..127
  const int acol = (tid & 1) * 16;  // 0 or 16
  const int bkrow = tid >> 3;       // 0..31 (non-BT path)
  const int bn16 = (tid & 7) * 16;  // 0..112

  floatx8 acc[4][2] = {};

  // stage A (TDM) / B tiles for K-step `step` into LDS buffer `buf`
  auto stage_tiles = [&](int step, int buf) {
    const int k0 = step * 32;
#if ATH_HAVE_TDM
    if (wave == 0) {
      tdm_load_tile_2d((unsigned)(unsigned long long)&sAf[buf][0][0],
                       A + (long)bm * K + k0, K, M, 32, 128, K);
      if (BT) {
        tdm_load_tile_2d((unsigned)(unsigned long long)&sBf[buf][0][0],
                         W + (long)bn * K + k0, K, N, 32, 128, K);
      }
    }
#else
    {  // manual A staging
      const float* pA = A + (long)(bm + arow) * K + k0 + acol;
#pragma unroll
      for (int i = 0; i < 16; i += 4)
        *(floatx4*)&sAf[buf][arow][acol + i] = *(const floatx4*)(pA + i);
    }
    if (BT) {
      int n = bn + arow;
      if (n < N) {
        const float* pW = W + (long)n * K + k0 + acol;
#pragma unroll
        for (int i = 0; i < 16; i += 4)
          *(floatx4*)&sBf[buf][arow][acol + i] = *(const floatx4*)(pW + i);
      } else {
#pragma unroll
        for (int i = 0; i < 16; ++i) sBf[buf][arow][acol + i] = 0.f;
      }
    }
#endif
    if (!BT) {  // manual transpose staging of B [K,N] -> LDS [n][k]
      const int gk = k0 + bkrow;
      const float* pW = W + (long)gk * N + bn + bn16;
      if (bn + 128 <= N) {
#pragma unroll
        for (int i = 0; i < 16; i += 4) {
          floatx4 f = *(const floatx4*)(pW + i);
          sBf[buf][bn16 + i + 0][bkrow] = f.x;
          sBf[buf][bn16 + i + 1][bkrow] = f.y;
          sBf[buf][bn16 + i + 2][bkrow] = f.z;
          sBf[buf][bn16 + i + 3][bkrow] = f.w;
        }
      } else {
#pragma unroll
        for (int i = 0; i < 16; ++i)
          sBf[buf][bn16 + i][bkrow] = (bn + bn16 + i < N) ? pW[i] : 0.f;
      }
    }
  };

  const int nsteps = K / 32;
  stage_tiles(0, 0);
#if ATH_HAVE_TDM
  if (wave == 0) __builtin_amdgcn_s_wait_tensorcnt(0);
#endif
  __syncthreads();

  for (int i = 0; i < nsteps; ++i) {
    const int cur = i & 1;
    if (i + 1 < nsteps) stage_tiles(i + 1, cur ^ 1);

    // ---- fragments per ISA 7.12.2 layouts (fp32 LDS -> bf16 regs) ----
    bf16x16 af[4];
#pragma unroll
    for (int mt = 0; mt < 4; ++mt) {
      const float* p = &sAf[cur][wm + mt * 16 + lrow][hf * 8];
      floatx4 f0 = *(const floatx4*)(p + 0);
      floatx4 f1 = *(const floatx4*)(p + 4);
      floatx4 f2 = *(const floatx4*)(p + 16);
      floatx4 f3 = *(const floatx4*)(p + 20);
      bf16x16 a;
#pragma unroll
      for (int j = 0; j < 4; ++j) {
        a[j + 0]  = to_bf16(f0[j]);
        a[j + 4]  = to_bf16(f1[j]);
        a[j + 8]  = to_bf16(f2[j]);
        a[j + 12] = to_bf16(f3[j]);
      }
      af[mt] = a;
    }
    bf16x16 bfr[2];
#pragma unroll
    for (int nt = 0; nt < 2; ++nt) {
      const float* p = &sBf[cur][wn + nt * 16 + lrow][hf * 8];
      floatx4 f0 = *(const floatx4*)(p + 0);
      floatx4 f1 = *(const floatx4*)(p + 4);
      floatx4 f2 = *(const floatx4*)(p + 16);
      floatx4 f3 = *(const floatx4*)(p + 20);
      bf16x16 b;
#pragma unroll
      for (int j = 0; j < 4; ++j) {
        b[j + 0]  = to_bf16(f0[j]);
        b[j + 4]  = to_bf16(f1[j]);
        b[j + 8]  = to_bf16(f2[j]);
        b[j + 12] = to_bf16(f3[j]);
      }
      bfr[nt] = b;
    }
#pragma unroll
    for (int mt = 0; mt < 4; ++mt)
#pragma unroll
      for (int nt = 0; nt < 2; ++nt)
        acc[mt][nt] = wmma_bf16(af[mt], bfr[nt], acc[mt][nt]);

#if ATH_HAVE_TDM
    if (wave == 0) __builtin_amdgcn_s_wait_tensorcnt(0);
#endif
    __syncthreads();
  }

  // ---- epilogue ----
#pragma unroll
  for (int mt = 0; mt < 4; ++mt) {
#pragma unroll
    for (int nt = 0; nt < 2; ++nt) {
      int gm = bm + wm + mt * 16 + hf * 8;
      int gn = bn + wn + nt * 16 + lrow;
      if (gn < N) {
        float bval = HAS_BIAS ? bias[gn] : 0.f;
#pragma unroll
        for (int i = 0; i < 8; ++i) {
          float v = acc[mt][nt][i] + bval;
          long idx = (long)(gm + i) * N + gn;
          if (HAS_RES) v += Res[idx];
          C[idx] = v;
        }
      }
    }
  }
}

// ---------------------------------------------------------------------------
// Flash-style causal attention with WMMA, GQA-blocked.
// grid = (S/128, KVH, B): each workgroup stages a K/V chunk once and runs
// BOTH sibling query heads (n_rep = 2) against it, halving K/V traffic.
// block = 256 (8 waves); wave w owns 16 query rows.
// ---------------------------------------------------------------------------
__global__ __launch_bounds__(256) void attn_kernel(
    const float* __restrict__ q, const float* __restrict__ k,
    const float* __restrict__ v, float* __restrict__ o, int S, int H, int KVH) {
  const int NREP = 2;  // H / KVH
  const int D = H * 64;
  const int KVD = KVH * 64;
  const int b = blockIdx.z, kvh = blockIdx.y;
  const int h0 = kvh * NREP;
  const int qblk = blockIdx.x * 128;
  const int tid = threadIdx.x;
  const int wave = tid >> 5, lane = tid & 31;
  const int hf = lane >> 4, lrow = lane & 15;

  __shared__ __bf16 sK[32][64];     // [key][d]
  __shared__ __bf16 sVt[64][32];    // [d][key]
  __shared__ float  sS[8][16][32];  // per-wave score scratch (reused per head)
  __shared__ __bf16 sP[8][16][32];  // per-wave softmax probs
  __shared__ float  sFac[8][16];
  __shared__ float  sL[8][16];

  const float* qp = q + (long)b * S * D + (long)h0 * 64;
  const float* kp = k + (long)b * S * KVD + (long)kvh * 64;
  const float* vp = v + (long)b * S * KVD + (long)kvh * 64;
  float* op = o + (long)b * S * D + (long)h0 * 64;

  const int qrow0 = qblk + wave * 16;
  const float scale = 0.125f;  // 1/sqrt(64)

  // Q fragments for both query heads sharing this kv head (pre-scaled)
  bf16x16 qf[NREP][2];
#pragma unroll
  for (int hq = 0; hq < NREP; ++hq) {
    const float* pq = qp + (long)(qrow0 + lrow) * D + hq * 64;
#pragma unroll
    for (int c = 0; c < 2; ++c) {
      int base = c * 32 + hf * 16;
      bf16x16 f;
#pragma unroll
      for (int j = 0; j < 8; ++j) f[j] = to_bf16(pq[base + j] * scale);
#pragma unroll
      for (int j = 0; j < 8; ++j) f[8 + j] = to_bf16(pq[base + 8 + j] * scale);
      qf[hq][c] = f;
    }
  }

  floatx8 oacc[NREP][4] = {};
  float m_i[NREP] = {-1e30f, -1e30f};
  float l_i[NREP] = {0.f, 0.f};  // lanes 0..15 own row = lane

  const int kend = qblk + 128;
  for (int kb = 0; kb < kend; kb += 32) {
    __syncthreads();
    {  // stage K and V^T chunk (32 keys x 64) -- once for both heads
      int r = tid >> 3;
      int c0 = (tid & 7) * 8;
      const float* pk = kp + (long)(kb + r) * KVD + c0;
      const float* pv = vp + (long)(kb + r) * KVD + c0;
#pragma unroll
      for (int j = 0; j < 8; ++j) sK[r][c0 + j] = to_bf16(pk[j]);
#pragma unroll
      for (int j = 0; j < 8; ++j) sVt[c0 + j][r] = to_bf16(pv[j]);
    }
    __syncthreads();

    if (kb <= qrow0 + 15) {  // wave-uniform: EXEC all-1 inside
#pragma unroll
      for (int hq = 0; hq < NREP; ++hq) {
        floatx8 sacc[2] = {};
#pragma unroll
        for (int kt = 0; kt < 2; ++kt) {
#pragma unroll
          for (int c = 0; c < 2; ++c) {
            bf16x16 kf = *(const bf16x16*)&sK[kt * 16 + lrow][c * 32 + hf * 16];
            sacc[kt] = wmma_bf16(qf[hq][c], kf, sacc[kt]);
          }
        }
#pragma unroll
        for (int kt = 0; kt < 2; ++kt) {
          int key = kb + kt * 16 + lrow;
#pragma unroll
          for (int i = 0; i < 8; ++i) {
            int qr = qrow0 + i + hf * 8;
            sS[wave][i + hf * 8][kt * 16 + lrow] =
                (key <= qr) ? sacc[kt][i] : -1e30f;
          }
        }
        if (lane < 16) {
          float mx = m_i[hq];
#pragma unroll 8
          for (int j = 0; j < 32; ++j) mx = fmaxf(mx, sS[wave][lane][j]);
          float fac = __expf(m_i[hq] - mx);
          float sum = 0.f;
#pragma unroll 8
          for (int j = 0; j < 32; ++j) {
            float e = __expf(sS[wave][lane][j] - mx);
            sP[wave][lane][j] = to_bf16(e);
            sum += e;
          }
          l_i[hq] = l_i[hq] * fac + sum;
          m_i[hq] = mx;
          sFac[wave][lane] = fac;
        }
        float facv[8];
#pragma unroll
        for (int i = 0; i < 8; ++i) facv[i] = sFac[wave][i + hf * 8];
        const __bf16* pp = &sP[wave][lrow][hf * 16];
        bf16x8 plo = *(const bf16x8*)pp;
        bf16x8 phi = *(const bf16x8*)(pp + 8);
        bf16x16 pf = __builtin_shufflevector(plo, phi, 0, 1, 2, 3, 4, 5, 6, 7,
                                             8, 9, 10, 11, 12, 13, 14, 15);
#pragma unroll
        for (int nt = 0; nt < 4; ++nt) {
          floatx8 t = oacc[hq][nt];
#pragma unroll
          for (int i = 0; i < 8; ++i) t[i] *= facv[i];
          bf16x16 vf = *(const bf16x16*)&sVt[nt * 16 + lrow][hf * 16];
          oacc[hq][nt] = wmma_bf16(pf, vf, t);
        }
      }
    }
  }

#pragma unroll
  for (int hq = 0; hq < NREP; ++hq) {
    if (lane < 16) sL[wave][lane] = l_i[hq];
    float lv[8];
#pragma unroll
    for (int i = 0; i < 8; ++i) {
      float d = sL[wave][i + hf * 8];
      lv[i] = (d > 0.f) ? 1.f / d : 0.f;
    }
#pragma unroll
    for (int nt = 0; nt < 4; ++nt) {
      int d = hq * 64 + nt * 16 + lrow;
#pragma unroll
      for (int i = 0; i < 8; ++i) {
        int row = qrow0 + i + hf * 8;
        op[(long)row * D + d] = oacc[hq][nt][i] * lv[i];
      }
    }
  }
}

// ---------------------------------------------------------------------------
// Host-side orchestration
// ---------------------------------------------------------------------------
extern "C" void kernel_launch(void* const* d_in, const int* in_sizes, int n_in,
                              void* d_out, int out_size, void* d_ws,
                              size_t ws_size, hipStream_t stream) {
  (void)in_sizes; (void)n_in; (void)out_size; (void)ws_size;
  const int B = 2, S = 2048, D = 1024, H = 16, KVH = 8, L = 4, Fd = 2752,
            V = 32000;
  const int M = B * S;
  const int KVD = KVH * 64;

  const int*   tokens = (const int*)d_in[0];
  const float* embed  = (const float*)d_in[1];
  const float* wq = (const float*)d_in[2];
  const float* bq = (const float*)d_in[3];
  const float* wk = (const float*)d_in[4];
  const float* bk = (const float*)d_in[5];
  const float* wv = (const float*)d_in[6];
  const float* bv = (const float*)d_in[7];
  const float* wo = (const float*)d_in[8];
  const float* bo = (const float*)d_in[9];
  const float* w1 = (const float*)d_in[10];
  const float* b1 = (const float*)d_in[11];
  const float* w2 = (const float*)d_in[12];
  const float* b2 = (const float*)d_in[13];
  const float* w3 = (const float*)d_in[14];
  const float* b3 = (const float*)d_in[15];
  const float* g1 = (const float*)d_in[16];
  const float* g2 = (const float*)d_in[17];
  const float* gpost = (const float*)d_in[18];
  const float* fcos = (const float*)d_in[19];
  const float* fsin = (const float*)d_in[20];

  char* w = (char*)d_ws;
  float* x    = (float*)w; w += (size_t)M * D * 4;
  float* xalt = (float*)w; w += (size_t)M * D * 4;
  float* hbuf = (float*)w; w += (size_t)M * D * 4;
  float* qbuf = (float*)w; w += (size_t)M * D * 4;
  float* kbuf = (float*)w; w += (size_t)M * KVD * 4;
  float* vbuf = (float*)w; w += (size_t)M * KVD * 4;
  float* obuf = (float*)w; w += (size_t)M * D * 4;
  float* u1   = (float*)w; w += (size_t)M * Fd * 4;
  float* u2   = (float*)w; w += (size_t)M * Fd * 4;

  dim3 blk(256);

  embed_gather_kernel<<<dim3((M * D + 255) / 256), blk, 0, stream>>>(
      tokens, embed, x, M, D);

  const dim3 gD(D / 128, M / 128);
  const dim3 gKV(KVD / 128, M / 128);
  const dim3 gF((Fd + 127) / 128, M / 128);
  const dim3 gV(V / 128, M / 128);

  for (int l = 0; l < L; ++l) {
    const float* Wq = wq + (size_t)l * D * D;
    const float* Bq = bq + (size_t)l * D;
    const float* Wk = wk + (size_t)l * D * KVD;
    const float* Bk = bk + (size_t)l * KVD;
    const float* Wv = wv + (size_t)l * D * KVD;
    const float* Bv = bv + (size_t)l * KVD;
    const float* Wo = wo + (size_t)l * D * D;
    const float* Bo = bo + (size_t)l * D;
    const float* W1 = w1 + (size_t)l * D * Fd;
    const float* B1 = b1 + (size_t)l * Fd;
    const float* W2 = w2 + (size_t)l * D * Fd;
    const float* B2 = b2 + (size_t)l * Fd;
    const float* W3 = w3 + (size_t)l * Fd * D;
    const float* B3 = b3 + (size_t)l * D;
    const float* G1 = g1 + (size_t)l * D;
    const float* G2 = g2 + (size_t)l * D;

    rmsnorm_kernel<<<dim3(M), blk, 0, stream>>>(x, G1, hbuf, D);
    gemm_wmma_kernel<false, true, false><<<gD, blk, 0, stream>>>(
        hbuf, Wq, Bq, nullptr, qbuf, M, D, D);
    gemm_wmma_kernel<false, true, false><<<gKV, blk, 0, stream>>>(
        hbuf, Wk, Bk, nullptr, kbuf, M, KVD, D);
    gemm_wmma_kernel<false, true, false><<<gKV, blk, 0, stream>>>(
        hbuf, Wv, Bv, nullptr, vbuf, M, KVD, D);
    rope_kernel<<<dim3((M * H * 32 + 255) / 256), blk, 0, stream>>>(
        qbuf, fcos, fsin, S, H, M * H * 32);
    rope_kernel<<<dim3((M * KVH * 32 + 255) / 256), blk, 0, stream>>>(
        kbuf, fcos, fsin, S, KVH, M * KVH * 32);
    attn_kernel<<<dim3(S / 128, KVH, B), blk, 0, stream>>>(qbuf, kbuf, vbuf,
                                                           obuf, S, H, KVH);
    gemm_wmma_kernel<false, true, true><<<gD, blk, 0, stream>>>(
        obuf, Wo, Bo, x, xalt, M, D, D);
    rmsnorm_kernel<<<dim3(M), blk, 0, stream>>>(xalt, G2, hbuf, D);
    gemm_wmma_kernel<false, true, false><<<gF, blk, 0, stream>>>(
        hbuf, W1, B1, nullptr, u1, M, Fd, D);
    gemm_wmma_kernel<false, true, false><<<gF, blk, 0, stream>>>(
        hbuf, W2, B2, nullptr, u2, M, Fd, D);
    silu_mul_kernel<<<dim3((M * Fd + 255) / 256), blk, 0, stream>>>(u1, u2,
                                                                    M * Fd);
    gemm_wmma_kernel<false, true, true><<<gD, blk, 0, stream>>>(
        u1, W3, B3, xalt, x, M, D, Fd);
  }

  rmsnorm_kernel<<<dim3(M), blk, 0, stream>>>(x, gpost, hbuf, D);
  gemm_wmma_kernel<true, false, false><<<gV, blk, 0, stream>>>(
      hbuf, embed, nullptr, nullptr, (float*)d_out, M, V, D);
}
